// MLLA_15564961480939
// MI455X (gfx1250) — compile-verified, matrix operand order checked by reading, not run on hardware
//
#include <hip/hip_runtime.h>
#include <hip/hip_bf16.h>

#define DEVI __device__ __forceinline__

typedef __attribute__((ext_vector_type(16))) __bf16 v16bf;
typedef __attribute__((ext_vector_type(8)))  float  v8f;

union FragBF {
  v16bf v;
  uint4 u[2];
  unsigned short s[16];
};

constexpr int Bc = 32, Hc = 56, Wc = 56, Cc = 192, NHc = 6, HDc = 32;
constexpr int Nc = Hc * Wc;          // 3136
constexpr int Mtok = Bc * Nc;        // 100352
constexpr int MLP = 4 * Cc;          // 768
constexpr int KV_CHUNKS = 7;         // 98 K-iterations / 7 = 14 per chunk

DEVI unsigned short f2bf(float f) {
  unsigned int u = __builtin_bit_cast(unsigned int, f);
  u = u + 0x7FFFu + ((u >> 16) & 1u);          // round-to-nearest-even
  return (unsigned short)(u >> 16);
}
DEVI float elu1(float x) { return x > 0.f ? x + 1.f : __expf(x); }
DEVI float siluf(float x) { return x / (1.f + __expf(-x)); }
DEVI float geluf(float x) { return 0.5f * x * (1.f + erff(x * 0.70710678118654752f)); }

// ---------------- fp32 -> bf16 weight conversion ----------------
__global__ void k_f32_to_bf16(const float* __restrict__ src,
                              unsigned short* __restrict__ dst, int n) {
  int t = blockIdx.x * blockDim.x + threadIdx.x;
  if (t < n) dst[t] = f2bf(src[t]);
}

// ---------------- depthwise 3x3 conv on (B,N,C) token layout ----------------
template <bool RES, bool SILU, bool BF16OUT>
__global__ void k_dwconv(const float* __restrict__ x, const float* __restrict__ w,
                         const float* __restrict__ bias, float* __restrict__ out,
                         unsigned short* __restrict__ out16) {
  int t = blockIdx.x * 256 + threadIdx.x;           // over B*N*C (fits int)
  int c = t % Cc;
  int n = (t / Cc) % Nc;
  int b = t / (Cc * Nc);
  int i = n / Wc, j = n % Wc;
  float acc = bias[c];
  const size_t bbase = (size_t)b * Nc * Cc;
#pragma unroll
  for (int di = 0; di < 3; di++) {
    int ii = i + di - 1;
    if (ii < 0 || ii >= Hc) continue;
#pragma unroll
    for (int dj = 0; dj < 3; dj++) {
      int jj = j + dj - 1;
      if (jj < 0 || jj >= Wc) continue;
      acc += w[c * 9 + di * 3 + dj] * x[bbase + (size_t)(ii * Wc + jj) * Cc + c];
    }
  }
  if (RES) acc += x[t];
  if (SILU) acc = siluf(acc);
  out[t] = acc;
  if (BF16OUT) out16[t] = f2bf(acc);
}

// ---------------- wave-per-token LayerNorm -> bf16 ----------------
__global__ void k_ln_bf16(const float* __restrict__ x, const float* __restrict__ g,
                          const float* __restrict__ be, unsigned short* __restrict__ out) {
  int wave = threadIdx.x >> 5, lane = threadIdx.x & 31;
  int tok = blockIdx.x * 8 + wave;
  const float* xp = x + (size_t)tok * Cc;
  float v[6];
  float s = 0.f, q = 0.f;
#pragma unroll
  for (int j = 0; j < 6; j++) {
    v[j] = xp[j * 32 + lane];
    s += v[j];
    q += v[j] * v[j];
  }
#pragma unroll
  for (int m = 16; m >= 1; m >>= 1) {
    s += __shfl_xor(s, m, 32);
    q += __shfl_xor(q, m, 32);
  }
  float mean = s * (1.f / Cc);
  float var = q * (1.f / Cc) - mean * mean;
  float rs = rsqrtf(var + 1e-5f);
  unsigned short* op = out + (size_t)tok * Cc;
#pragma unroll
  for (int j = 0; j < 6; j++) {
    int c = j * 32 + lane;
    op[c] = f2bf((v[j] - mean) * rs * g[c] + be[c]);
  }
}

// ---------------- WMMA GEMM: out = act(A[M,K] @ W[K,N] + bias) (+res) ----------------
// block = 256 threads = 8 waves; wave computes 16(M) x 64(N); grid (M/128, N/64).
// W panel (256K x 64N) staged transposed in LDS so B fragments are two 16B ds reads.
template <int ACT, bool RES>
__global__ void __launch_bounds__(256)
k_gemm(const unsigned short* __restrict__ A, const unsigned short* __restrict__ Wt,
       const float* __restrict__ bias, const float* __restrict__ res,
       float* __restrict__ out32, unsigned short* __restrict__ out16, int Nn, int K) {
  __shared__ unsigned short ldsW[64 * 256];
  const int tid = threadIdx.x;
  const int wave = tid >> 5, lane = tid & 31;
  const int half = lane >> 4, l16 = lane & 15;
  const int nblk = blockIdx.y * 64;
  const int mwave = (blockIdx.x * 8 + wave) * 16;

  v8f acc[4];
#pragma unroll
  for (int t = 0; t < 4; t++)
#pragma unroll
    for (int r = 0; r < 8; r++) acc[t][r] = 0.f;

  const int arow = mwave + l16;
  for (int kb = 0; kb < K; kb += 256) {
    const int kc = (K - kb) < 256 ? (K - kb) : 256;
    // prefetch next K-chunk of A while this panel is being staged/consumed
    if (kb + 256 < K)
      __builtin_prefetch(A + (size_t)arow * K + kb + 256, 0, 1);
    __syncthreads();
    for (int idx = tid; idx < 64 * kc; idx += 256) {
      int kk = idx / 64, nn = idx % 64;
      ldsW[nn * 256 + kk] = Wt[(size_t)(kb + kk) * Nn + nblk + nn];
    }
    __syncthreads();
    for (int k0 = 0; k0 < kc; k0 += 32) {
      FragBF a;
      const unsigned short* ap = A + (size_t)arow * K + (kb + k0) + half * 8;
      a.u[0] = *(const uint4*)(ap);
      a.u[1] = *(const uint4*)(ap + 16);
#pragma unroll
      for (int nt = 0; nt < 4; nt++) {
        FragBF bf;
        const unsigned short* wp = &ldsW[(nt * 16 + l16) * 256 + k0 + half * 8];
        bf.u[0] = *(const uint4*)(wp);
        bf.u[1] = *(const uint4*)(wp + 16);
        acc[nt] = __builtin_amdgcn_wmma_f32_16x16x32_bf16(false, a.v, false, bf.v,
                                                          (short)0, acc[nt], false, false);
      }
    }
  }
#pragma unroll
  for (int nt = 0; nt < 4; nt++) {
    int gn = nblk + nt * 16 + l16;
    float bb = bias[gn];
#pragma unroll
    for (int r = 0; r < 8; r++) {
      int gm = mwave + half * 8 + r;
      float v = acc[nt][r] + bb;
      if (ACT == 1) v = siluf(v);
      if (ACT == 2) v = geluf(v);
      if (RES) v += res[(size_t)gm * Nn + gn];
      size_t o = (size_t)gm * Nn + gn;
      if (out32) out32[o] = v;
      if (out16) out16[o] = f2bf(v);
    }
  }
}

// ---------------- k_mean over N per (b,head,d) ----------------
__global__ void k_kmean(const float* __restrict__ qk, float* __restrict__ km) {
  int bhd = blockIdx.x;
  int d = bhd % HDc;
  int head = (bhd / HDc) % NHc;
  int b = bhd / (HDc * NHc);
  int col = Cc + head * HDc + d;
  float s = 0.f;
  for (int n = threadIdx.x; n < Nc; n += 256)
    s += elu1(qk[((size_t)b * Nc + n) * (2 * Cc) + col]);
  __shared__ float red[256];
  red[threadIdx.x] = s;
  __syncthreads();
  for (int st = 128; st > 0; st >>= 1) {
    if (threadIdx.x < st) red[threadIdx.x] += red[threadIdx.x + st];
    __syncthreads();
  }
  if (threadIdx.x == 0) km[bhd] = red[0] * (1.f / Nc);
}

// ---------------- elu+1 + RoPE -> bf16 q_rope/k_rope in (b,h,n,d) ----------------
__global__ void k_rope(const float* __restrict__ qk, unsigned short* __restrict__ qr,
                       unsigned short* __restrict__ kr) {
  int t = blockIdx.x * 256 + threadIdx.x;   // over B*N*96 pairs
  int p = t % 96;
  int n = (t / 96) % Nc;
  int b = t / (96 * Nc);
  int i = n / Wc, j = n % Wc;
  int kk = p < 48 ? p : p - 48;
  float th = __powf(10000.f, -(float)kk * (1.f / 48.f));
  float ang = (p < 48 ? (float)i : (float)j) * th;
  float cs = __cosf(ang), sn = __sinf(ang);
  size_t base = ((size_t)b * Nc + n) * (2 * Cc);
  float q0 = elu1(qk[base + 2 * p]), q1 = elu1(qk[base + 2 * p + 1]);
  float k0 = elu1(qk[base + Cc + 2 * p]), k1 = elu1(qk[base + Cc + 2 * p + 1]);
  int head = (2 * p) / HDc, d = (2 * p) % HDc;
  size_t o = ((size_t)(b * NHc + head) * Nc + n) * HDc + d;
  qr[o] = f2bf(cs * q0 - sn * q1);
  qr[o + 1] = f2bf(sn * q0 + cs * q1);
  kr[o] = f2bf(cs * k0 - sn * k1);
  kr[o + 1] = f2bf(sn * k0 + cs * k1);
}

// ---------------- z = 1/(q . k_mean + eps) ----------------
__global__ void k_z(const float* __restrict__ qk, const float* __restrict__ km,
                    float* __restrict__ z) {
  int t = blockIdx.x * 256 + threadIdx.x;   // t = (b*NH+head)*N + n
  int n = t % Nc;
  int head = (t / Nc) % NHc;
  int b = t / (Nc * NHc);
  const float* qp = qk + ((size_t)b * Nc + n) * (2 * Cc) + head * HDc;
  const float* kp = km + (b * NHc + head) * HDc;
  float s = 0.f;
#pragma unroll
  for (int d = 0; d < HDc; d++) s += elu1(qp[d]) * kp[d];
  z[t] = 1.f / (s + 1e-6f);
}

// ---------------- kv partials: chunk c of k_rope^T @ v  (32x32 via WMMA) ----------------
// grid (B*NH, KV_CHUNKS), 1 wave per block; deterministic split-N reduction (no atomics).
__global__ void __launch_bounds__(32)
k_kv(const unsigned short* __restrict__ kr, const unsigned short* __restrict__ vb,
     float* __restrict__ kvpart) {
  int bh = blockIdx.x;
  int chunk = blockIdx.y;
  int b = bh / NHc, head = bh % NHc;
  int lane = threadIdx.x & 31;
  int half = lane >> 4, l16 = lane & 15;
  v8f acc[4];
#pragma unroll
  for (int t = 0; t < 4; t++)
#pragma unroll
    for (int r = 0; r < 8; r++) acc[t][r] = 0.f;
  const unsigned short* krb = kr + (size_t)bh * Nc * HDc;
  const unsigned short* vbb = vb + (size_t)b * Nc * Cc + head * HDc;
  const int nbeg = chunk * (Nc / KV_CHUNKS);        // 448 per chunk, 14 iters of 32
  const int nend = nbeg + (Nc / KV_CHUNKS);
  for (int n0 = nbeg; n0 < nend; n0 += 32) {
    FragBF a[2], bf[2];
#pragma unroll
    for (int dmi = 0; dmi < 2; dmi++) {
      int d = dmi * 16 + l16;
#pragma unroll
      for (int e = 0; e < 8; e++) {
        int na = n0 + half * 8 + e;
        int nb2 = n0 + 16 + half * 8 + e;
        a[dmi].s[e] = krb[(size_t)na * HDc + d];
        a[dmi].s[8 + e] = krb[(size_t)nb2 * HDc + d];
      }
    }
#pragma unroll
    for (int emi = 0; emi < 2; emi++) {
      int e2 = emi * 16 + l16;
#pragma unroll
      for (int e = 0; e < 8; e++) {
        int na = n0 + half * 8 + e;
        int nb2 = n0 + 16 + half * 8 + e;
        bf[emi].s[e] = vbb[(size_t)na * Cc + e2];
        bf[emi].s[8 + e] = vbb[(size_t)nb2 * Cc + e2];
      }
    }
#pragma unroll
    for (int dmi = 0; dmi < 2; dmi++)
#pragma unroll
      for (int emi = 0; emi < 2; emi++)
        acc[dmi * 2 + emi] = __builtin_amdgcn_wmma_f32_16x16x32_bf16(
            false, a[dmi].v, false, bf[emi].v, (short)0, acc[dmi * 2 + emi], false, false);
  }
  float* op = kvpart + ((size_t)chunk * (Bc * NHc) + bh) * (HDc * HDc);
#pragma unroll
  for (int dmi = 0; dmi < 2; dmi++)
#pragma unroll
    for (int emi = 0; emi < 2; emi++)
#pragma unroll
      for (int r = 0; r < 8; r++) {
        int row = dmi * 16 + half * 8 + r;
        int col = emi * 16 + l16;
        op[row * HDc + col] = acc[dmi * 2 + emi][r];
      }
}

// ---------------- reduce kv partials, scale 1/N, -> bf16 ----------------
__global__ void k_kvfin(const float* __restrict__ kvpart, unsigned short* __restrict__ kvb) {
  int t = blockIdx.x * 256 + threadIdx.x;           // over B*NH*HD*HD = 196608
  float s = 0.f;
#pragma unroll
  for (int c = 0; c < KV_CHUNKS; c++)
    s += kvpart[(size_t)c * (Bc * NHc * HDc * HDc) + t];
  kvb[t] = f2bf(s * (1.f / (float)Nc));
}

// ---------------- out = (q_rope @ kv) * z  -> (b,n,c) fp32 ----------------
__global__ void __launch_bounds__(256)
k_attn(const unsigned short* __restrict__ qr, const unsigned short* __restrict__ kvb,
       const float* __restrict__ z, float* __restrict__ out) {
  int gw = blockIdx.x * 8 + (threadIdx.x >> 5);
  int lane = threadIdx.x & 31;
  int half = lane >> 4, l16 = lane & 15;
  int ntile = gw % (Nc / 16);
  int head = (gw / (Nc / 16)) % NHc;
  int b = gw / ((Nc / 16) * NHc);
  int bh = b * NHc + head;
  int n0 = ntile * 16;
  FragBF a;
  const unsigned short* qp = qr + ((size_t)bh * Nc + n0 + l16) * HDc + half * 8;
  a.u[0] = *(const uint4*)(qp);
  a.u[1] = *(const uint4*)(qp + 16);
  const unsigned short* kp = kvb + (size_t)bh * HDc * HDc;
  v8f acc[2];
#pragma unroll
  for (int t = 0; t < 2; t++)
#pragma unroll
    for (int r = 0; r < 8; r++) acc[t][r] = 0.f;
#pragma unroll
  for (int emi = 0; emi < 2; emi++) {
    FragBF bf;
    int col = emi * 16 + l16;
#pragma unroll
    for (int e = 0; e < 8; e++) {
      bf.s[e] = kp[(half * 8 + e) * HDc + col];
      bf.s[8 + e] = kp[(16 + half * 8 + e) * HDc + col];
    }
    acc[emi] = __builtin_amdgcn_wmma_f32_16x16x32_bf16(false, a.v, false, bf.v,
                                                       (short)0, acc[emi], false, false);
  }
#pragma unroll
  for (int r = 0; r < 8; r++) {
    int nr = n0 + half * 8 + r;
    float zv = z[(size_t)bh * Nc + nr];
#pragma unroll
    for (int emi = 0; emi < 2; emi++) {
      int c = head * HDc + emi * 16 + l16;
      out[((size_t)b * Nc + nr) * Cc + c] = acc[emi][r] * zv;
    }
  }
}

// ---------------- y = (attn + lepe) * act_res -> bf16 ----------------
__global__ void k_y(const float* __restrict__ attn, const float* __restrict__ lepe,
                    const float* __restrict__ act, unsigned short* __restrict__ y) {
  int t = blockIdx.x * 256 + threadIdx.x;
  y[t] = f2bf((attn[t] + lepe[t]) * act[t]);
}

// ================= host side =================
extern "C" void kernel_launch(void* const* d_in, const int* in_sizes, int n_in,
                              void* d_out, int out_size, void* d_ws, size_t ws_size,
                              hipStream_t stream) {
  const float* x = (const float*)d_in[0];
  const float* cpe1w = (const float*)d_in[1];
  const float* cpe1b = (const float*)d_in[2];
  const float* ln1g = (const float*)d_in[3];
  const float* ln1b = (const float*)d_in[4];
  const float* inw = (const float*)d_in[5];
  const float* inb = (const float*)d_in[6];
  const float* actw = (const float*)d_in[7];
  const float* actb = (const float*)d_in[8];
  const float* dwcw = (const float*)d_in[9];
  const float* dwcb = (const float*)d_in[10];
  const float* qkw = (const float*)d_in[11];
  const float* qkb = (const float*)d_in[12];
  const float* lepew = (const float*)d_in[13];
  const float* lepeb = (const float*)d_in[14];
  const float* outw = (const float*)d_in[15];
  const float* outb = (const float*)d_in[16];
  const float* cpe2w = (const float*)d_in[17];
  const float* cpe2b = (const float*)d_in[18];
  const float* ln2g = (const float*)d_in[19];
  const float* ln2b = (const float*)d_in[20];
  const float* fc1w = (const float*)d_in[21];
  const float* fc1b = (const float*)d_in[22];
  const float* fc2w = (const float*)d_in[23];
  const float* fc2b = (const float*)d_in[24];

  char* ws = (char*)d_ws;

  constexpr size_t BNC = (size_t)Bc * Nc * Cc;        // 19,267,584
  constexpr size_t F32S = BNC * 4;
  constexpr size_t BF16S = BNC * 2;

  // small region
  constexpr size_t OFF_W_IN = 0;
  constexpr size_t OFF_W_ACT = OFF_W_IN + (size_t)Cc * Cc * 2;
  constexpr size_t OFF_W_QK = OFF_W_ACT + (size_t)Cc * Cc * 2;
  constexpr size_t OFF_W_OUT = OFF_W_QK + (size_t)Cc * 2 * Cc * 2;
  constexpr size_t OFF_W_FC1 = OFF_W_OUT + (size_t)Cc * Cc * 2;
  constexpr size_t OFF_W_FC2 = OFF_W_FC1 + (size_t)Cc * MLP * 2;
  constexpr size_t OFF_KMEAN = OFF_W_FC2 + (size_t)MLP * Cc * 2;
  constexpr size_t OFF_KV = OFF_KMEAN + (size_t)Bc * NHc * HDc * 4;
  constexpr size_t OFF_Z = OFF_KV + (size_t)Bc * NHc * HDc * HDc * 2;
  constexpr size_t OFF_KVP = OFF_Z + (size_t)Bc * NHc * Nc * 4;
  // big region (16MB aligned start; kv partials end ~9.3MB)
  constexpr size_t OFF_A = 16777216;                // x after cpe1 (shortcut); later x2
  constexpr size_t OFF_C = OFF_A + F32S;            // act_res f32
  constexpr size_t OFF_D = OFF_C + F32S;            // h f32 -> attn_out f32 -> (with E) m1_bf16
  constexpr size_t OFF_E = OFF_D + F32S;            // v f32 (h after dwconv+silu)
  constexpr size_t OFF_G = OFF_E + F32S;            // qk f32 (2*F32S) -> lepe (lo) + x_mid (hi)
  constexpr size_t OFF_Bb = OFF_G + 2 * F32S;       // bf16: xn1 -> y -> xn2
  constexpr size_t OFF_F = OFF_Bb + BF16S;          // h bf16 (v bf16)
  constexpr size_t OFF_Hq = OFF_F + BF16S;          // q_rope bf16
  constexpr size_t OFF_Hk = OFF_Hq + BF16S;         // k_rope bf16

  unsigned short* Wb_in = (unsigned short*)(ws + OFF_W_IN);
  unsigned short* Wb_act = (unsigned short*)(ws + OFF_W_ACT);
  unsigned short* Wb_qk = (unsigned short*)(ws + OFF_W_QK);
  unsigned short* Wb_out = (unsigned short*)(ws + OFF_W_OUT);
  unsigned short* Wb_fc1 = (unsigned short*)(ws + OFF_W_FC1);
  unsigned short* Wb_fc2 = (unsigned short*)(ws + OFF_W_FC2);
  float* kmean = (float*)(ws + OFF_KMEAN);
  unsigned short* kvb = (unsigned short*)(ws + OFF_KV);
  float* zbuf = (float*)(ws + OFF_Z);
  float* kvpart = (float*)(ws + OFF_KVP);
  float* Abuf = (float*)(ws + OFF_A);
  float* Cbuf = (float*)(ws + OFF_C);
  float* Dbuf = (float*)(ws + OFF_D);
  float* Ebuf = (float*)(ws + OFF_E);
  float* qkbuf = (float*)(ws + OFF_G);
  float* lepe = (float*)(ws + OFF_G);               // reuse after qk consumed
  float* xmid = (float*)(ws + OFF_G + F32S);
  unsigned short* bnc16 = (unsigned short*)(ws + OFF_Bb);
  unsigned short* h16 = (unsigned short*)(ws + OFF_F);
  unsigned short* qr16 = (unsigned short*)(ws + OFF_Hq);
  unsigned short* kr16 = (unsigned short*)(ws + OFF_Hk);
  unsigned short* m1_16 = (unsigned short*)(ws + OFF_D);   // 2*F32S span over D+E
  float* outp = (float*)d_out;

  const int ELT_BLK = (int)(BNC / 256);             // 75264 for B*N*C grids

  // 0) weights -> bf16
  k_f32_to_bf16<<<(Cc * Cc + 255) / 256, 256, 0, stream>>>(inw, Wb_in, Cc * Cc);
  k_f32_to_bf16<<<(Cc * Cc + 255) / 256, 256, 0, stream>>>(actw, Wb_act, Cc * Cc);
  k_f32_to_bf16<<<(Cc * 2 * Cc + 255) / 256, 256, 0, stream>>>(qkw, Wb_qk, Cc * 2 * Cc);
  k_f32_to_bf16<<<(Cc * Cc + 255) / 256, 256, 0, stream>>>(outw, Wb_out, Cc * Cc);
  k_f32_to_bf16<<<(Cc * MLP + 255) / 256, 256, 0, stream>>>(fc1w, Wb_fc1, Cc * MLP);
  k_f32_to_bf16<<<(MLP * Cc + 255) / 256, 256, 0, stream>>>(fc2w, Wb_fc2, MLP * Cc);

  // 1) cpe1: A = x + dwconv(x)
  k_dwconv<true, false, false><<<ELT_BLK, 256, 0, stream>>>(x, cpe1w, cpe1b, Abuf, nullptr);
  // 2) LN1 -> bf16
  k_ln_bf16<<<Mtok / 8, 256, 0, stream>>>(Abuf, ln1g, ln1b, bnc16);
  // 3) act_res = silu(xn @ act_w + b)
  k_gemm<1, false><<<dim3(Mtok / 128, Cc / 64), 256, 0, stream>>>(
      bnc16, Wb_act, actb, nullptr, Cbuf, nullptr, Cc, Cc);
  // 4) h = xn @ in_w + b
  k_gemm<0, false><<<dim3(Mtok / 128, Cc / 64), 256, 0, stream>>>(
      bnc16, Wb_in, inb, nullptr, Dbuf, nullptr, Cc, Cc);
  // 5) v = silu(dwconv(h) + b)  (fp32 + bf16)
  k_dwconv<false, true, true><<<ELT_BLK, 256, 0, stream>>>(Dbuf, dwcw, dwcb, Ebuf, h16);
  // 6) qk = v @ qk_w + b
  k_gemm<0, false><<<dim3(Mtok / 128, 2 * Cc / 64), 256, 0, stream>>>(
      h16, Wb_qk, qkb, nullptr, qkbuf, nullptr, 2 * Cc, Cc);
  // 7) rope + elu -> q_rope/k_rope bf16
  k_rope<<<(int)((size_t)Bc * Nc * 96 / 256), 256, 0, stream>>>(qkbuf, qr16, kr16);
  // 8) k_mean
  k_kmean<<<Bc * NHc * HDc, 256, 0, stream>>>(qkbuf, kmean);
  // 9) z
  k_z<<<Bc * NHc * Nc / 256, 256, 0, stream>>>(qkbuf, kmean, zbuf);
  // 10) kv partials (WMMA; (b,h) x 7 N-chunks) then deterministic reduce
  k_kv<<<dim3(Bc * NHc, KV_CHUNKS), 32, 0, stream>>>(kr16, h16, kvpart);
  k_kvfin<<<Bc * NHc * HDc * HDc / 256, 256, 0, stream>>>(kvpart, kvb);
  // 11) attn_out = (q_rope @ kv) * z  -> D
  k_attn<<<Bc * NHc * (Nc / 16) / 8, 256, 0, stream>>>(qr16, kvb, zbuf, Dbuf);
  // 12) lepe = dwconv(v) + b  (into low half of G; qk no longer needed)
  k_dwconv<false, false, false><<<ELT_BLK, 256, 0, stream>>>(Ebuf, lepew, lepeb, lepe, nullptr);
  // 13) y = (attn + lepe) * act_res -> bf16
  k_y<<<ELT_BLK, 256, 0, stream>>>(Dbuf, lepe, Cbuf, bnc16);
  // 14) x_mid = shortcut + y @ out_w + b
  k_gemm<0, true><<<dim3(Mtok / 128, Cc / 64), 256, 0, stream>>>(
      bnc16, Wb_out, outb, Abuf, xmid, nullptr, Cc, Cc);
  // 15) cpe2: x2 = x_mid + dwconv(x_mid)  -> A
  k_dwconv<true, false, false><<<ELT_BLK, 256, 0, stream>>>(xmid, cpe2w, cpe2b, Abuf, nullptr);
  // 16) LN2 -> bf16
  k_ln_bf16<<<Mtok / 8, 256, 0, stream>>>(Abuf, ln2g, ln2b, bnc16);
  // 17) m1 = gelu(xn2 @ fc1 + b) -> bf16 (overlays D+E)
  k_gemm<2, false><<<dim3(Mtok / 128, MLP / 64), 256, 0, stream>>>(
      bnc16, Wb_fc1, fc1b, nullptr, nullptr, m1_16, MLP, Cc);
  // 18) out = x2 + m1 @ fc2 + b
  k_gemm<0, true><<<dim3(Mtok / 128, Cc / 64), 256, 0, stream>>>(
      m1_16, Wb_fc2, fc2b, Abuf, outp, nullptr, Cc, MLP);
}
